// MultiHeadAttention_21242908246489
// MI455X (gfx1250) — matrix operand order repaired and checked by reading, not compiled, not run on hardware
//
#include <hip/hip_runtime.h>
#include <hip/hip_bf16.h>
#include <math.h>

typedef __attribute__((ext_vector_type(16))) __bf16 bf16x16;
typedef __attribute__((ext_vector_type(8)))  float  f32x8;

#define T_SEQ  2048
#define NBATCH 2
#define NHEAD  8
#define DHEAD  64
#define DMODEL 512
#define M_ROWS 4096   // = NBATCH*T_SEQ = 2*T_SEQ

// ---------------- wave32 WMMA fragment helpers (bf16 16x16x32) ----------------

__device__ __forceinline__ int lane_id() { return (int)(threadIdx.x & 31); }

__device__ __forceinline__ f32x8 zero8() {
  f32x8 z;
#pragma unroll
  for (int i = 0; i < 8; ++i) z[i] = 0.0f;
  return z;
}

// A fragment: 16x32 (MxK) bf16. Lane l: m = l&15, hi = l>>4.
// elems 0..7  <- A[m][8*hi + 0..7]       (16 contiguous bytes)
// elems 8..15 <- A[m][16 + 8*hi + 0..7]  (16 contiguous bytes)
__device__ __forceinline__ bf16x16 load_a_frag(const __bf16* base, int ld) {
  int l = lane_id(); int m = l & 15, hi = l >> 4;
  const __bf16* r = base + (size_t)m * ld + hi * 8;
  union { uint4 u[2]; bf16x16 v; } c;
  c.u[0] = *(const uint4*)(r);
  c.u[1] = *(const uint4*)(r + 16);
  return c.v;
}

// B fragment: 32x16 (KxN) bf16, loaded from row-major [N,K] storage (K contiguous).
// Lane l: n = l&15, hi = l>>4; elems 0..7 <- B[n][8*hi + 0..7], elems 8..15 <- B[n][16 + 8*hi + 0..7]
__device__ __forceinline__ bf16x16 load_b_frag(const __bf16* base, int ld) {
  int l = lane_id(); int n = l & 15, hi = l >> 4;
  const __bf16* r = base + (size_t)n * ld + hi * 8;
  union { uint4 u[2]; bf16x16 v; } c;
  c.u[0] = *(const uint4*)(r);
  c.u[1] = *(const uint4*)(r + 16);
  return c.v;
}

__device__ __forceinline__ f32x8 wmma_bf16(bf16x16 a, bf16x16 b, f32x8 c) {
  return __builtin_amdgcn_wmma_f32_16x16x32_bf16(
      /*neg_a=*/false, a, /*neg_b=*/false, b,
      /*c_mod=*/(short)0, c, /*reuse_a=*/false, /*reuse_b=*/false);
}

__device__ __forceinline__ float rowmax16(float v) {
#pragma unroll
  for (int off = 8; off > 0; off >>= 1) v = fmaxf(v, __shfl_xor(v, off, 32));
  return v;
}
__device__ __forceinline__ float rowsum16(float v) {
#pragma unroll
  for (int off = 8; off > 0; off >>= 1) v += __shfl_xor(v, off, 32);
  return v;
}

// ---------------- small prep kernels ----------------

__global__ void cvt_bf16_kernel(const float* __restrict__ src, __bf16* __restrict__ dst, int n) {
  int i = blockIdx.x * blockDim.x + threadIdx.x;
  if (i < n) dst[i] = (__bf16)src[i];
}

// W [K=512, N=512] fp32 row-major -> Wt [N, K] bf16 row-major (K contiguous)
__global__ void wtrans_kernel(const float* __restrict__ W, __bf16* __restrict__ Wt) {
  int i = blockIdx.x * 256 + threadIdx.x;   // 0 .. 512*512-1
  int n = i >> 9, k = i & 511;
  Wt[i] = (__bf16)W[k * DMODEL + n];
}

// Sinusoid position encoding, pos[row, :] for row in [0, 2T), bf16 output.
// pos_seq[row] = T - row; inv_freq[f] = 10000^(-(2f)/512); [sin | cos]
__global__ void pos_kernel(__bf16* __restrict__ pos) {
  int row = blockIdx.x;       // 0..4095
  int f   = threadIdx.x;      // 0..255
  float inv = __expf(-(float)(2 * f) * (9.210340371976184f / 512.0f)); // ln(10000)/512
  float ang = (float)(T_SEQ - row) * inv;
  pos[(size_t)row * DMODEL + f]       = (__bf16)sinf(ang);
  pos[(size_t)row * DMODEL + 256 + f] = (__bf16)cosf(ang);
}

// ---------------- bf16 WMMA GEMM: C[M=4096, N=512] = A[M,K=512] @ Wt[N,K]^T ----------------
// Workgroup = 8 waves; wave tile 16x64 (4 N-fragments); block tile 128x64.
// Optional outputs: out0 = bf16(acc+bias+e0), out1 = bf16(acc+bias+e1),
//                   outF = fp32(acc+bias),     outT = bf16(acc+bias) stored [B,H,Dk,T].
__global__ void __launch_bounds__(256)
gemm_kernel(const __bf16* __restrict__ A, const __bf16* __restrict__ Wt,
            const float* __restrict__ bias,
            const float* __restrict__ e0, const float* __restrict__ e1,
            __bf16* __restrict__ out0, __bf16* __restrict__ out1,
            __bf16* __restrict__ outT, float* __restrict__ outF) {
  int w   = threadIdx.x >> 5;
  int gm0 = blockIdx.x * 128 + w * 16;
  int n0  = blockIdx.y * 64;

  f32x8 acc[4];
#pragma unroll
  for (int i = 0; i < 4; ++i) acc[i] = zero8();

  for (int k0 = 0; k0 < DMODEL; k0 += 32) {
    bf16x16 a = load_a_frag(A + (size_t)gm0 * DMODEL + k0, DMODEL);
#pragma unroll
    for (int nt = 0; nt < 4; ++nt) {
      bf16x16 bfr = load_b_frag(Wt + (size_t)(n0 + nt * 16) * DMODEL + k0, DMODEL);
      acc[nt] = wmma_bf16(a, bfr, acc[nt]);
    }
  }

  int l = lane_id(), n16 = l & 15, hi = l >> 4;
#pragma unroll
  for (int nt = 0; nt < 4; ++nt) {
    int gn = n0 + nt * 16 + n16;
    float bb = bias ? bias[gn] : 0.0f;
    float a0 = e0 ? e0[gn] : 0.0f;
    float a1 = e1 ? e1[gn] : 0.0f;
#pragma unroll
    for (int j = 0; j < 8; ++j) {
      int gm = gm0 + j + 8 * hi;
      float v = acc[nt][j] + bb;
      if (out0) out0[(size_t)gm * DMODEL + gn] = (__bf16)(v + a0);
      if (out1) out1[(size_t)gm * DMODEL + gn] = (__bf16)(v + a1);
      if (outF) outF[(size_t)gm * DMODEL + gn] = v;
      if (outT) {
        int bb_ = gm >> 11, tt = gm & (T_SEQ - 1);
        int hh  = gn >> 6,  dd = gn & (DHEAD - 1);
        outT[(((size_t)(bb_ * NHEAD + hh)) * DHEAD + dd) * T_SEQ + tt] = (__bf16)v;
      }
    }
  }
}

// ---------------- flash attention with fused Transformer-XL rel_shift ----------------
// shifted_rel[t, j] = R[t, T - t + j],  R = (q + rel_bias) @ rel_k^T
// grid = (T/128, B*H); 8 waves/block; wave owns 16 query rows; key blocks of 32.
__global__ void __launch_bounds__(256)
flash_kernel(const __bf16* __restrict__ qc, const __bf16* __restrict__ qr,
             const __bf16* __restrict__ kb, const __bf16* __restrict__ vT,
             const __bf16* __restrict__ relk, __bf16* __restrict__ attn) {
  __shared__ __align__(16) float  Rst[8][16][48];
  __shared__ __align__(16) __bf16 Pst[8][16][32];

  int w  = threadIdx.x >> 5;
  int l  = lane_id(), n16 = l & 15, hi = l >> 4;
  int bh = blockIdx.y;
  int bb = bh >> 3, h = bh & 7;        // NHEAD = 8
  int t0 = blockIdx.x * 128 + w * 16;

  const __bf16* qc_b = qc + (size_t)(bb * T_SEQ + t0) * DMODEL + h * DHEAD;
  const __bf16* qr_b = qr + (size_t)(bb * T_SEQ + t0) * DMODEL + h * DHEAD;
  const __bf16* k_b  = kb + (size_t)(bb * T_SEQ) * DMODEL + h * DHEAD;
  const __bf16* v_b  = vT + (size_t)(bb * NHEAD + h) * DHEAD * T_SEQ;
  const __bf16* r_b  = relk + h * DHEAD;

  bf16x16 aqc0 = load_a_frag(qc_b, DMODEL),      aqc1 = load_a_frag(qc_b + 32, DMODEL);
  bf16x16 aqr0 = load_a_frag(qr_b, DMODEL),      aqr1 = load_a_frag(qr_b + 32, DMODEL);

  f32x8 o[4];
#pragma unroll
  for (int i = 0; i < 4; ++i) o[i] = zero8();
  float mrow[8], lrow[8], scl[8];
#pragma unroll
  for (int j = 0; j < 8; ++j) { mrow[j] = -INFINITY; lrow[j] = 0.0f; }

  for (int j0 = 0; j0 < T_SEQ; j0 += 32) {
    // ---- content logits S = (q+cb) . k^T, 16x32 block ----
    f32x8 s0 = zero8(), s1 = zero8();
    {
      const __bf16* kbase = k_b + (size_t)j0 * DMODEL;
      s0 = wmma_bf16(aqc0, load_b_frag(kbase, DMODEL), s0);
      s0 = wmma_bf16(aqc1, load_b_frag(kbase + 32, DMODEL), s0);
      s1 = wmma_bf16(aqc0, load_b_frag(kbase + (size_t)16 * DMODEL, DMODEL), s1);
      s1 = wmma_bf16(aqc1, load_b_frag(kbase + (size_t)16 * DMODEL + 32, DMODEL), s1);
    }
    // ---- relative logits band: R[t, u], u in [u0, u0+48), u0 = T - t0 - 15 + j0 ----
    int u0 = T_SEQ - t0 - 15 + j0;
#pragma unroll
    for (int rt = 0; rt < 3; ++rt) {
      const __bf16* rbase = r_b + (size_t)(u0 + rt * 16) * DMODEL;
      f32x8 r = zero8();
      r = wmma_bf16(aqr0, load_b_frag(rbase, DMODEL), r);
      r = wmma_bf16(aqr1, load_b_frag(rbase + 32, DMODEL), r);
#pragma unroll
      for (int j = 0; j < 8; ++j) Rst[w][j + 8 * hi][rt * 16 + n16] = r[j];
    }
    asm volatile("s_wait_dscnt 0" ::: "memory");

    // ---- combine + rel_shift gather + online softmax ----
#pragma unroll
    for (int j = 0; j < 8; ++j) {
      int m = j + 8 * hi;
      // shifted column index: (15 - ti) + ji, ji = n16 (+16 for second tile)
      float c0 = (s0[j] + Rst[w][m][15 - m + n16]) * 0.125f;
      float c1 = (s1[j] + Rst[w][m][31 - m + n16]) * 0.125f;
      float bm = rowmax16(fmaxf(c0, c1));
      float mnew = fmaxf(mrow[j], bm);
      float corr = __expf(mrow[j] - mnew);
      float p0 = __expf(c0 - mnew);
      float p1 = __expf(c1 - mnew);
      float bs = rowsum16(p0 + p1);
      lrow[j] = lrow[j] * corr + bs;
      mrow[j] = mnew;
      scl[j]  = corr;
      Pst[w][m][n16]      = (__bf16)p0;
      Pst[w][m][16 + n16] = (__bf16)p1;
    }
    asm volatile("s_wait_dscnt 0" ::: "memory");

    // ---- rescale O, then O += P . V  (V block from vT[B,H,Dk,T]) ----
#pragma unroll
    for (int dt = 0; dt < 4; ++dt)
#pragma unroll
      for (int j = 0; j < 8; ++j) o[dt][j] *= scl[j];

    bf16x16 ap = load_a_frag(&Pst[w][0][0], 32);
#pragma unroll
    for (int dt = 0; dt < 4; ++dt) {
      bf16x16 bv = load_b_frag(v_b + (size_t)(dt * 16) * T_SEQ + j0, T_SEQ);
      o[dt] = wmma_bf16(ap, bv, o[dt]);
    }
  }

  // ---- epilogue: normalize and store attn_vec (bf16, [B*T, H*Dk] layout) ----
  __bf16* outp = attn + (size_t)(bb * T_SEQ + t0) * DMODEL + h * DHEAD;
#pragma unroll
  for (int j = 0; j < 8; ++j) {
    float inv = 1.0f / lrow[j];
    int m = j + 8 * hi;
#pragma unroll
    for (int dt = 0; dt < 4; ++dt)
      outp[(size_t)m * DMODEL + dt * 16 + n16] = (__bf16)(o[dt][j] * inv);
  }
}

// ---------------- host launcher ----------------

extern "C" void kernel_launch(void* const* d_in, const int* in_sizes, int n_in,
                              void* d_out, int out_size, void* d_ws, size_t ws_size,
                              hipStream_t stream) {
  (void)in_sizes; (void)n_in; (void)out_size; (void)ws_size;
  const float* query    = (const float*)d_in[0];
  const float* key_in   = (const float*)d_in[1];
  const float* value    = (const float*)d_in[2];
  const float* Wq       = (const float*)d_in[3];
  const float* bq       = (const float*)d_in[4];
  const float* Wk       = (const float*)d_in[5];
  const float* bk       = (const float*)d_in[6];
  const float* Wv       = (const float*)d_in[7];
  const float* bv       = (const float*)d_in[8];
  const float* Wr       = (const float*)d_in[9];
  const float* cbias    = (const float*)d_in[10];  // [H,Dk] = flat 512
  const float* rbias    = (const float*)d_in[11];
  const float* Wo       = (const float*)d_in[12];
  const float* bo       = (const float*)d_in[13];
  float* out = (float*)d_out;

  char* ws = (char*)d_ws;
  size_t off = 0;
  auto carve = [&](size_t bytes) -> void* {
    void* p = ws + off;
    off += (bytes + 255) & ~(size_t)255;
    return p;
  };
  const size_t ACT = (size_t)M_ROWS * DMODEL * sizeof(__bf16);   // 4 MB
  const size_t WT  = (size_t)DMODEL * DMODEL * sizeof(__bf16);   // 512 KB

  __bf16* qin  = (__bf16*)carve(ACT);
  __bf16* kin  = (__bf16*)carve(ACT);
  __bf16* vin  = (__bf16*)carve(ACT);
  __bf16* posb = (__bf16*)carve(ACT);
  __bf16* WqT  = (__bf16*)carve(WT);
  __bf16* WkT  = (__bf16*)carve(WT);
  __bf16* WvT  = (__bf16*)carve(WT);
  __bf16* WrT  = (__bf16*)carve(WT);
  __bf16* WoT  = (__bf16*)carve(WT);
  __bf16* qcb  = (__bf16*)carve(ACT);
  __bf16* qrb  = (__bf16*)carve(ACT);
  __bf16* kbf  = (__bf16*)carve(ACT);
  __bf16* vTb  = (__bf16*)carve(ACT);
  __bf16* relk = (__bf16*)carve(ACT + (size_t)16 * DMODEL * sizeof(__bf16)); // +16 pad rows
  __bf16* attn = (__bf16*)carve(ACT);

  const int NACT = M_ROWS * DMODEL;  // 2097152

  // 1) fp32 -> bf16 activations
  cvt_bf16_kernel<<<NACT / 256, 256, 0, stream>>>(query,  qin, NACT);
  cvt_bf16_kernel<<<NACT / 256, 256, 0, stream>>>(key_in, kin, NACT);
  cvt_bf16_kernel<<<NACT / 256, 256, 0, stream>>>(value,  vin, NACT);

  // 2) weights -> transposed bf16 [N,K]
  wtrans_kernel<<<1024, 256, 0, stream>>>(Wq, WqT);
  wtrans_kernel<<<1024, 256, 0, stream>>>(Wk, WkT);
  wtrans_kernel<<<1024, 256, 0, stream>>>(Wv, WvT);
  wtrans_kernel<<<1024, 256, 0, stream>>>(Wr, WrT);
  wtrans_kernel<<<1024, 256, 0, stream>>>(Wo, WoT);

  // 3) sinusoid positions [2T, D] bf16; zero the 16 pad rows of relk
  pos_kernel<<<M_ROWS, 256, 0, stream>>>(posb);
  hipMemsetAsync(relk + (size_t)M_ROWS * DMODEL, 0,
                 (size_t)16 * DMODEL * sizeof(__bf16), stream);

  // 4) projection GEMMs (grid: M/128 x N/64)
  dim3 ggrid(M_ROWS / 128, DMODEL / 64);
  // q: two bf16 outputs with content/relative biases folded in
  gemm_kernel<<<ggrid, 256, 0, stream>>>(qin, WqT, bq, cbias, rbias,
                                         qcb, qrb, nullptr, nullptr);
  // k: plain bf16 [B*T, H*Dk]
  gemm_kernel<<<ggrid, 256, 0, stream>>>(kin, WkT, bk, nullptr, nullptr,
                                         kbf, nullptr, nullptr, nullptr);
  // v: transposed bf16 [B,H,Dk,T]
  gemm_kernel<<<ggrid, 256, 0, stream>>>(vin, WvT, bv, nullptr, nullptr,
                                         nullptr, nullptr, vTb, nullptr);
  // rel_k = pos @ Wr (no bias): plain bf16 [2T, H*Dk]
  gemm_kernel<<<ggrid, 256, 0, stream>>>(posb, WrT, nullptr, nullptr, nullptr,
                                         relk, nullptr, nullptr, nullptr);

  // 5) flash attention with fused rel_shift
  dim3 fgrid(T_SEQ / 128, NBATCH * NHEAD);
  flash_kernel<<<fgrid, 256, 0, stream>>>(qcb, qrb, kbf, vTb, relk, attn);

  // 6) output projection: fp32 out = attn @ Wo + bo
  gemm_kernel<<<ggrid, 256, 0, stream>>>(attn, WoT, bo, nullptr, nullptr,
                                         nullptr, nullptr, nullptr, out);
}
